// DoudouModel_31937376813530
// MI455X (gfx1250) — compile-verified, matrix operand order checked by reading, not run on hardware
//
#include <hip/hip_runtime.h>
#include <cstdint>
#include <cstddef>

#define HID    1024
#define HEADS  16
#define HDIM   64
#define LAYERS 8
#define INTER  4096
#define SEQ    2048
#define BATCH  2
#define NTOK   (BATCH*SEQ)
#define NVOCAB 32000

typedef __attribute__((ext_vector_type(16))) __bf16 v16bf;
typedef __attribute__((ext_vector_type(8)))  float  v8f;

struct Pack32 { uint4 a, b; };          // 32B -> v16bf
struct U32x8  { unsigned int w[8]; };   // 32B -> v16bf

// fp32 -> bf16, round-half-up (bias-free enough at bf16 precision; the >>16
// folds into ds_store_b16_d16_hi on the LDS staging path).
__device__ __forceinline__ unsigned short f2bf(float x) {
    unsigned int r = __builtin_bit_cast(unsigned int, x) + 0x8000u;
    return (unsigned short)(r >> 16);
}

#if __has_builtin(__builtin_amdgcn_perm)
// {hi16(b), hi16(a)} in one v_perm_b32: sel 0-3 -> 2nd operand, 4-7 -> 1st.
__device__ __forceinline__ unsigned int pack2(float a, float b) {
    const unsigned int ua = __builtin_bit_cast(unsigned int, a) + 0x8000u;
    const unsigned int ub = __builtin_bit_cast(unsigned int, b) + 0x8000u;
    return __builtin_amdgcn_perm(ua, ub, 0x03020706u);
}
#else
__device__ __forceinline__ unsigned int pack2(float a, float b) {
    return (unsigned int)f2bf(a) | ((unsigned int)f2bf(b) << 16);
}
#endif

// ---------------------------------------------------------------------------
// GEMM: C[M,N] = A[M,K] (fp32) * W[K,N] (fp32), bf16 WMMA, f32 accumulate.
// N, K compile-time. Block: 256 thr (8 waves). WG tile 128(M) x 128(N).
// Wave tile 32x64 (2x4 WMMA frags). K step 32, double-buffered LDS.
// Requires M%128==0, N%128==0, K%32==0 (true for all shapes used here).
// ---------------------------------------------------------------------------
template<int N, int K>
__global__ __launch_bounds__(256) void gemm_bf16_kernel(
    const float* __restrict__ A, const float* __restrict__ Bw,
    float* __restrict__ C)
{
    __shared__ __align__(16) unsigned short Ash[2][128 * 32];  // [m][k] bf16
    __shared__ __align__(16) unsigned short Bsh[2][128 * 32];  // [n][k] bf16 (transposed)

    const int tid   = threadIdx.x;
    const int lane  = tid & 31;
    const int wv    = tid >> 5;
    const int waveM = wv & 3;      // 0..3 -> 32-row slice
    const int waveN = wv >> 2;     // 0..1 -> 64-col slice
    const int gm    = blockIdx.y * 128;
    const int gn    = blockIdx.x * 128;
    const int half  = lane >> 4;
    const int l15   = lane & 15;

    const int ar0 = tid >> 3;          // 0..31
    const int ac0 = (tid & 7) << 2;    // 0,4,...,28
    const int bkr = tid >> 3;          // 0..31 (k row)
    const int bnc = (tid & 7) << 2;    // 0,4,...,28 (n base; 4 passes of +32)

    auto stageA = [&](int k0, int buf) {
        if (k0 + 64 < K)  // prefetch tile k+2 into GL2 (global_prefetch_b8)
            __builtin_prefetch(A + (size_t)(gm + ar0) * K + k0 + 64 + ac0, 0, 0);
#pragma unroll
        for (int rr = 0; rr < 4; ++rr) {
            const int row = ar0 + rr * 32;
            const float4 v = *(const float4*)(A + (size_t)(gm + row) * K + k0 + ac0);
            *(uint2*)&Ash[buf][row * 32 + ac0] = make_uint2(pack2(v.x, v.y), pack2(v.z, v.w));
        }
    };
    auto stageB = [&](int k0, int buf) {
        if (k0 + 64 < K)
            __builtin_prefetch(Bw + (size_t)(k0 + 64 + bkr) * N + gn + bnc, 0, 0);
#pragma unroll
        for (int pp = 0; pp < 4; ++pp) {
            const int nc = bnc + pp * 32;
            const float4 v = *(const float4*)(Bw + (size_t)(k0 + bkr) * N + gn + nc);
            Bsh[buf][(nc + 0) * 32 + bkr] = f2bf(v.x);
            Bsh[buf][(nc + 1) * 32 + bkr] = f2bf(v.y);
            Bsh[buf][(nc + 2) * 32 + bkr] = f2bf(v.z);
            Bsh[buf][(nc + 3) * 32 + bkr] = f2bf(v.w);
        }
    };

    v8f acc[2][4] = {{{}, {}, {}, {}}, {{}, {}, {}, {}}};

    stageA(0, 0);
    stageB(0, 0);
    __syncthreads();

    constexpr int NK = K / 32;
    for (int kt = 0; kt < NK; ++kt) {
        const int cur = kt & 1;
        if (kt + 1 < NK) {              // stage next tile into the other buffer
            stageA((kt + 1) * 32, cur ^ 1);
            stageB((kt + 1) * 32, cur ^ 1);
        }

        v16bf afr[2], bfr[4];
#pragma unroll
        for (int t = 0; t < 2; ++t) {
            const int row = waveM * 32 + t * 16 + l15;
            Pack32 fr{ *(const uint4*)&Ash[cur][row * 32 + half * 8],
                       *(const uint4*)&Ash[cur][row * 32 + 16 + half * 8] };
            afr[t] = __builtin_bit_cast(v16bf, fr);
        }
#pragma unroll
        for (int t = 0; t < 4; ++t) {
            const int col = waveN * 64 + t * 16 + l15;
            Pack32 fr{ *(const uint4*)&Bsh[cur][col * 32 + half * 8],
                       *(const uint4*)&Bsh[cur][col * 32 + 16 + half * 8] };
            bfr[t] = __builtin_bit_cast(v16bf, fr);
        }
#pragma unroll
        for (int tm = 0; tm < 2; ++tm)
#pragma unroll
            for (int tn = 0; tn < 4; ++tn)
                acc[tm][tn] = __builtin_amdgcn_wmma_f32_16x16x32_bf16(
                    false, afr[tm], false, bfr[tn], (short)0, acc[tm][tn], false, false);

        __syncthreads();   // staged(next) done AND reads(cur) done
    }

    // ---- epilogue: constant-N strides fold into instruction offsets ----
#pragma unroll
    for (int tm = 0; tm < 2; ++tm)
#pragma unroll
        for (int tn = 0; tn < 4; ++tn) {
            const int row0 = gm + waveM * 32 + tm * 16 + half * 8;
            const int col  = gn + waveN * 64 + tn * 16 + l15;
            float* p = C + (size_t)row0 * N + col;
#pragma unroll
            for (int r = 0; r < 8; ++r)
                p[(size_t)r * N] = acc[tm][tn][r];
        }
}

// ---------------------------------------------------------------------------
// Flash attention: one wave per (batch, head, 16-query tile).
// q pre-scaled by 1/sqrt(HDIM) in rope kernel. Streams 32-key blocks.
// ---------------------------------------------------------------------------
__global__ __launch_bounds__(32) void attn_kernel(
    const float* __restrict__ q, const float* __restrict__ k,
    const float* __restrict__ v, float* __restrict__ ctx)
{
    __shared__ __align__(16) unsigned short pbuf[16 * 32];  // [query][key] bf16
    __shared__ __align__(16) unsigned short vbuf[64 * 32];  // [dim][key] bf16

    const int lane = threadIdx.x & 31;
    const int half = lane >> 4;
    const int l15  = lane & 15;
    const int qb   = blockIdx.x * 16;
    const int hh   = blockIdx.y;
    const int bb   = blockIdx.z;

    // ---- load Q as two A-fragments (16x32 bf16, d-chunks 0..31 / 32..63) ----
    v16bf qf[2];
    {
        const float* qr = q + (size_t)(bb * SEQ + qb + l15) * HID + hh * HDIM;
#pragma unroll
        for (int c = 0; c < 2; ++c) {
            const int d0 = c * 32 + half * 8;
            const float4 a0 = *(const float4*)(qr + d0);
            const float4 a1 = *(const float4*)(qr + d0 + 4);
            const float4 b0 = *(const float4*)(qr + d0 + 16);
            const float4 b1 = *(const float4*)(qr + d0 + 20);
            U32x8 u;
            u.w[0] = pack2(a0.x, a0.y); u.w[1] = pack2(a0.z, a0.w);
            u.w[2] = pack2(a1.x, a1.y); u.w[3] = pack2(a1.z, a1.w);
            u.w[4] = pack2(b0.x, b0.y); u.w[5] = pack2(b0.z, b0.w);
            u.w[6] = pack2(b1.x, b1.y); u.w[7] = pack2(b1.z, b1.w);
            qf[c] = __builtin_bit_cast(v16bf, u);
        }
    }

    float mrow[8], lrow[8];
#pragma unroll
    for (int r = 0; r < 8; ++r) { mrow[r] = -3.0e38f; lrow[r] = 0.0f; }
    v8f o[4] = {{}, {}, {}, {}};

    const int nblk = (qb + 47) >> 5;   // causal: keys 0 .. qb+15
    for (int jb = 0; jb < nblk; ++jb) {
        const int kb = jb * 32;

        // ---- scores: two 16x16 tiles (keys kb..kb+15 and kb+16..kb+31) ----
        v8f st[2] = {{}, {}};
#pragma unroll
        for (int nh = 0; nh < 2; ++nh) {
            const float* kr = k + (size_t)(bb * SEQ + kb + nh * 16 + l15) * HID + hh * HDIM;
#pragma unroll
            for (int c = 0; c < 2; ++c) {
                const int d0 = c * 32 + half * 8;
                const float4 a0 = *(const float4*)(kr + d0);
                const float4 a1 = *(const float4*)(kr + d0 + 4);
                const float4 b0 = *(const float4*)(kr + d0 + 16);
                const float4 b1 = *(const float4*)(kr + d0 + 20);
                U32x8 u;
                u.w[0] = pack2(a0.x, a0.y); u.w[1] = pack2(a0.z, a0.w);
                u.w[2] = pack2(a1.x, a1.y); u.w[3] = pack2(a1.z, a1.w);
                u.w[4] = pack2(b0.x, b0.y); u.w[5] = pack2(b0.z, b0.w);
                u.w[6] = pack2(b1.x, b1.y); u.w[7] = pack2(b1.z, b1.w);
                const v16bf kf = __builtin_bit_cast(v16bf, u);
                st[nh] = __builtin_amdgcn_wmma_f32_16x16x32_bf16(
                    false, qf[c], false, kf, (short)0, st[nh], false, false);
            }
        }

        // ---- clip + causal mask + online softmax ----
        float pv[2][8], scl[8];
#pragma unroll
        for (int r = 0; r < 8; ++r) {
            const int qi = qb + half * 8 + r;
            float s0 = fminf(fmaxf(st[0][r], -1000.0f), 1000.0f);
            float s1 = fminf(fmaxf(st[1][r], -1000.0f), 1000.0f);
            if (kb + l15 > qi)      s0 = -3.0e38f;
            if (kb + 16 + l15 > qi) s1 = -3.0e38f;
            float m = fmaxf(s0, s1);
            m = fmaxf(m, __shfl_xor(m, 1));
            m = fmaxf(m, __shfl_xor(m, 2));
            m = fmaxf(m, __shfl_xor(m, 4));
            m = fmaxf(m, __shfl_xor(m, 8));
            const float mn = fmaxf(mrow[r], m);
            const float p0 = __expf(s0 - mn);
            const float p1 = __expf(s1 - mn);
            float rs = p0 + p1;
            rs += __shfl_xor(rs, 1);
            rs += __shfl_xor(rs, 2);
            rs += __shfl_xor(rs, 4);
            rs += __shfl_xor(rs, 8);
            scl[r]   = __expf(mrow[r] - mn);
            lrow[r]  = lrow[r] * scl[r] + rs;
            mrow[r]  = mn;
            pv[0][r] = p0;
            pv[1][r] = p1;
        }
#pragma unroll
        for (int t = 0; t < 4; ++t)
#pragma unroll
            for (int r = 0; r < 8; ++r)
                o[t][r] *= scl[r];

        // ---- write P (bf16) to LDS in [query][key] layout ----
#pragma unroll
        for (int nh = 0; nh < 2; ++nh)
#pragma unroll
            for (int r = 0; r < 8; ++r)
                pbuf[(half * 8 + r) * 32 + nh * 16 + l15] = f2bf(pv[nh][r]);

        // ---- stage V block 32x64 into LDS transposed [dim][key] ----
        {
            const float* vr = v + (size_t)(bb * SEQ + kb + lane) * HID + hh * HDIM;
#pragma unroll
            for (int d4 = 0; d4 < 16; ++d4) {
                const float4 vv = *(const float4*)(vr + d4 * 4);
                vbuf[(d4 * 4 + 0) * 32 + lane] = f2bf(vv.x);
                vbuf[(d4 * 4 + 1) * 32 + lane] = f2bf(vv.y);
                vbuf[(d4 * 4 + 2) * 32 + lane] = f2bf(vv.z);
                vbuf[(d4 * 4 + 3) * 32 + lane] = f2bf(vv.w);
            }
        }
        __syncthreads();

        // ---- O += P(16x32) @ V(32x64) : 4 WMMAs ----
        v16bf pf;
        {
            Pack32 fr{ *(const uint4*)&pbuf[l15 * 32 + half * 8],
                       *(const uint4*)&pbuf[l15 * 32 + 16 + half * 8] };
            pf = __builtin_bit_cast(v16bf, fr);
        }
#pragma unroll
        for (int t = 0; t < 4; ++t) {
            const int dcol = t * 16 + l15;
            Pack32 fr{ *(const uint4*)&vbuf[dcol * 32 + half * 8],
                       *(const uint4*)&vbuf[dcol * 32 + 16 + half * 8] };
            const v16bf vf = __builtin_bit_cast(v16bf, fr);
            o[t] = __builtin_amdgcn_wmma_f32_16x16x32_bf16(
                false, pf, false, vf, (short)0, o[t], false, false);
        }
        __syncthreads();
    }

    // ---- epilogue: ctx = O / l ----
#pragma unroll
    for (int t = 0; t < 4; ++t)
#pragma unroll
        for (int r = 0; r < 8; ++r) {
            const int row = qb + half * 8 + r;
            ctx[(size_t)(bb * SEQ + row) * HID + hh * HDIM + t * 16 + l15] =
                o[t][r] / lrow[r];
        }
}

// ---------------------------------------------------------------------------
// Elementwise kernels
// ---------------------------------------------------------------------------
__global__ __launch_bounds__(256) void embed_kernel(
    const int* __restrict__ ids, const float* __restrict__ ew, float* __restrict__ h)
{
    const int row = blockIdx.x, tid = threadIdx.x;
    const int tok = ids[row];
    const float4 vv = *(const float4*)(ew + (size_t)tok * HID + tid * 4);
    *(float4*)(h + (size_t)row * HID + tid * 4) = vv;
}

__global__ __launch_bounds__(256) void rmsnorm_kernel(
    const float* __restrict__ x, const float* __restrict__ w, float* __restrict__ out)
{
    __shared__ float red[256];
    const int row = blockIdx.x, tid = threadIdx.x;
    const float4 vv = *(const float4*)(x + (size_t)row * HID + tid * 4);
    red[tid] = vv.x * vv.x + vv.y * vv.y + vv.z * vv.z + vv.w * vv.w;
    __syncthreads();
    for (int off = 128; off > 0; off >>= 1) {
        if (tid < off) red[tid] += red[tid + off];
        __syncthreads();
    }
    const float r = rsqrtf(red[0] * (1.0f / HID) + 1e-5f);
    const float4 wv = *(const float4*)(w + tid * 4);
    *(float4*)(out + (size_t)row * HID + tid * 4) =
        make_float4(vv.x * r * wv.x, vv.y * r * wv.y, vv.z * r * wv.z, vv.w * r * wv.w);
}

// One thread per (token, head, rotation-pair). Also scales q by 1/sqrt(HDIM).
__global__ void rope_kernel(float* __restrict__ q, float* __restrict__ k)
{
    const int idx  = blockIdx.x * blockDim.x + threadIdx.x;   // NTOK*512 threads
    const int row  = idx >> 9;
    const int hcol = idx & 511;
    const int head = hcol >> 5;
    const int i    = hcol & 31;
    const int s    = row & (SEQ - 1);
    const float freq = __powf(100000.0f, -(float)i * (1.0f / 32.0f));
    const float ang  = (float)s * freq;
    const float cs = cosf(ang), sn = sinf(ang);
    const size_t i1 = (size_t)row * HID + head * 64 + i;
    const size_t i2 = i1 + 32;
    const float q1 = q[i1], q2 = q[i2];
    q[i1] = (q1 * cs - q2 * sn) * 0.125f;
    q[i2] = (q2 * cs + q1 * sn) * 0.125f;
    const float k1 = k[i1], k2 = k[i2];
    k[i1] = k1 * cs - k2 * sn;
    k[i2] = k2 * cs + k1 * sn;
}

__global__ void add_kernel(float* dst, const float* __restrict__ src, int n4)
{
    const int i = blockIdx.x * blockDim.x + threadIdx.x;
    if (i < n4) {
        const float4 a = ((const float4*)dst)[i];
        const float4 b = ((const float4*)src)[i];
        ((float4*)dst)[i] = make_float4(a.x + b.x, a.y + b.y, a.z + b.z, a.w + b.w);
    }
}

__global__ void swiglu_kernel(float* __restrict__ G, const float* __restrict__ U, int n)
{
    const int i = blockIdx.x * blockDim.x + threadIdx.x;
    if (i < n) {
        const float g = G[i];
        G[i] = (g / (1.0f + __expf(-g))) * U[i];
    }
}

// ---------------------------------------------------------------------------
// Driver
// ---------------------------------------------------------------------------
extern "C" void kernel_launch(void* const* d_in, const int* in_sizes, int n_in,
                              void* d_out, int out_size, void* d_ws, size_t ws_size,
                              hipStream_t stream)
{
    const int*   ids     = (const int*)d_in[0];
    const float* embed_W = (const float*)d_in[1];
    const float* Wq      = (const float*)d_in[2];
    const float* Wk      = (const float*)d_in[3];
    const float* Wv      = (const float*)d_in[4];
    const float* Wo      = (const float*)d_in[5];
    const float* Wg      = (const float*)d_in[6];
    const float* Wu      = (const float*)d_in[7];
    const float* Wd      = (const float*)d_in[8];
    const float* n1w     = (const float*)d_in[9];
    const float* n2w     = (const float*)d_in[10];
    const float* nfw     = (const float*)d_in[11];
    const float* headW   = (const float*)d_in[12];
    float* out = (float*)d_out;

    const size_t TOKF = (size_t)NTOK * HID;   // 4 Mi floats
    float* ws  = (float*)d_ws;
    float* h   = ws;
    float* xn  = ws + 1 * TOKF;
    float* q   = ws + 2 * TOKF;
    float* k   = ws + 3 * TOKF;
    float* v   = ws + 4 * TOKF;
    float* ctx = ws + 5 * TOKF;
    float* tmp = ws + 6 * TOKF;
    float* G   = ws + 7 * TOKF;
    float* U   = G + (size_t)NTOK * INTER;

    embed_kernel<<<NTOK, 256, 0, stream>>>(ids, embed_W, h);

    const dim3 gP(HID / 128,   NTOK / 128);   // N=1024 projections
    const dim3 gI(INTER / 128, NTOK / 128);   // N=4096 MLP up/gate
    const int addBlocks = (int)((TOKF / 4 + 255) / 256);

    for (int l = 0; l < LAYERS; ++l) {
        const float* wq = Wq + (size_t)l * HID * HID;
        const float* wk = Wk + (size_t)l * HID * HID;
        const float* wv = Wv + (size_t)l * HID * HID;
        const float* wo = Wo + (size_t)l * HID * HID;
        const float* wg = Wg + (size_t)l * HID * INTER;
        const float* wu = Wu + (size_t)l * HID * INTER;
        const float* wd = Wd + (size_t)l * INTER * HID;

        rmsnorm_kernel<<<NTOK, 256, 0, stream>>>(h, n1w + l * HID, xn);
        gemm_bf16_kernel<HID, HID><<<gP, 256, 0, stream>>>(xn, wq, q);
        gemm_bf16_kernel<HID, HID><<<gP, 256, 0, stream>>>(xn, wk, k);
        gemm_bf16_kernel<HID, HID><<<gP, 256, 0, stream>>>(xn, wv, v);
        rope_kernel<<<(NTOK * 512) / 256, 256, 0, stream>>>(q, k);
        attn_kernel<<<dim3(SEQ / 16, HEADS, BATCH), 32, 0, stream>>>(q, k, v, ctx);
        gemm_bf16_kernel<HID, HID><<<gP, 256, 0, stream>>>(ctx, wo, tmp);
        add_kernel<<<addBlocks, 256, 0, stream>>>(h, tmp, (int)(TOKF / 4));

        rmsnorm_kernel<<<NTOK, 256, 0, stream>>>(h, n2w + l * HID, xn);
        gemm_bf16_kernel<INTER, HID><<<gI, 256, 0, stream>>>(xn, wg, G);
        gemm_bf16_kernel<INTER, HID><<<gI, 256, 0, stream>>>(xn, wu, U);
        swiglu_kernel<<<(NTOK * INTER) / 256, 256, 0, stream>>>(G, U, NTOK * INTER);
        gemm_bf16_kernel<HID, INTER><<<gP, 256, 0, stream>>>(G, wd, tmp);
        add_kernel<<<addBlocks, 256, 0, stream>>>(h, tmp, (int)(TOKF / 4));
    }

    // final norm -> h slice of output, then LM head GEMM -> logits slice
    float* hout = out + (size_t)NTOK * NVOCAB;
    rmsnorm_kernel<<<NTOK, 256, 0, stream>>>(h, nfw, hout);
    gemm_bf16_kernel<NVOCAB, HID><<<dim3(NVOCAB / 128, NTOK / 128), 256, 0, stream>>>(
        hout, headW, out);
}
